// Lfm2ShortConv_37374805409986
// MI455X (gfx1250) — matrix operand order repaired and checked by reading, not compiled, not run on hardware
//
#include <hip/hip_runtime.h>

// ---------------------------------------------------------------------------
// LFM2 short-conv block on gfx1250 (MI455X, wave32, WMMA):
//   1) f32 -> bf16 conversion of hidden_states, W_in, W_out   (ws)
//   2) GEMM1: BCx = hidden @ W_in^T   (bf16 x bf16 -> f32)    v_wmma_f32_16x16x32_bf16
//   3) gate + depthwise causal conv(L=3) + gate -> y (bf16)
//   4) GEMM2: out = y @ W_out^T       (bf16 x bf16 -> f32)
// ---------------------------------------------------------------------------

typedef __attribute__((ext_vector_type(16))) __bf16 v16bf;
typedef __attribute__((ext_vector_type(8)))  __bf16 v8bf;
typedef __attribute__((ext_vector_type(4)))  __bf16 v4bf;
typedef __attribute__((ext_vector_type(8)))  float  v8f;
typedef __attribute__((ext_vector_type(4)))  float  v4f;

union FragAB {
    v16bf v;
    v8bf  h[2];
};

// ----------------------------- f32 -> bf16 ---------------------------------
__global__ __launch_bounds__(256)
void cvt_f32_to_bf16_x4(const float* __restrict__ in, __bf16* __restrict__ out, int n4) {
    int i = blockIdx.x * blockDim.x + threadIdx.x;
    if (i >= n4) return;
    v4f x = ((const v4f*)in)[i];
    v4bf o;
    o.x = (__bf16)x.x;
    o.y = (__bf16)x.y;
    o.z = (__bf16)x.z;
    o.w = (__bf16)x.w;
    ((v4bf*)out)[i] = o;
}

// ------------------------------ WMMA GEMM ----------------------------------
// C[M,N] = A[M,K] * B[N,K]^T, all row-major; M % 128 == 0, N % 256 == 0, K % 32 == 0.
// 256 threads = 8 waves arranged 2 (M) x 4 (N); each wave owns a 64x64 tile
// (4x4 grid of 16x16 WMMA accumulators).
__global__ __launch_bounds__(256)
void gemm_bf16_wmma(const __bf16* __restrict__ A,
                    const __bf16* __restrict__ B,
                    float* __restrict__ C,
                    int K, int N)
{
    const int tid  = threadIdx.x;
    const int wave = tid >> 5;
    const int lane = tid & 31;
    const int l    = lane & 15;
    const int hi   = lane >> 4;

    const int waveM = wave & 1;
    const int waveN = wave >> 1;

    const int m0 = blockIdx.x * 128 + waveM * 64;
    const int n0 = blockIdx.y * 256 + waveN * 64;

    const __bf16* Arow[4];
    const __bf16* Brow[4];
#pragma unroll
    for (int i = 0; i < 4; ++i)
        Arow[i] = A + (size_t)(m0 + i * 16 + l) * (size_t)K;
#pragma unroll
    for (int j = 0; j < 4; ++j)
        Brow[j] = B + (size_t)(n0 + j * 16 + l) * (size_t)K;

    const v8f zero = {0.f, 0.f, 0.f, 0.f, 0.f, 0.f, 0.f, 0.f};
    v8f acc[4][4];
#pragma unroll
    for (int i = 0; i < 4; ++i)
#pragma unroll
        for (int j = 0; j < 4; ++j)
            acc[i][j] = zero;

    // Per-lane K offsets dictated by the 16-bit WMMA VGPR layouts:
    //   A (16x32): lanes 0-15 hold K {kb..kb+7, kb+16..kb+23}; lanes 16-31 shift +8.
    //   B (32x16): lanes 0-15 hold K {kb..kb+7, kb+16..kb+23}; lanes 16-31 shift +8.
    const int aoff = hi * 8;
    const int boff = hi * 8;

    for (int kb = 0; kb < K; kb += 32) {
        FragAB a[4], b[4];
#pragma unroll
        for (int i = 0; i < 4; ++i) {
            a[i].h[0] = *(const v8bf*)(Arow[i] + kb + aoff);
            a[i].h[1] = *(const v8bf*)(Arow[i] + kb + aoff + 16);
        }
#pragma unroll
        for (int j = 0; j < 4; ++j) {
            b[j].h[0] = *(const v8bf*)(Brow[j] + kb + boff);
            b[j].h[1] = *(const v8bf*)(Brow[j] + kb + boff + 16);
        }
#pragma unroll
        for (int i = 0; i < 4; ++i)
#pragma unroll
            for (int j = 0; j < 4; ++j)
                acc[i][j] = __builtin_amdgcn_wmma_f32_16x16x32_bf16(
                    false, a[i].v, false, b[j].v, (short)0, acc[i][j], false, false);
    }

    // C/D layout: VGPR r, lanes 0-15 -> (M = r, N = l); lanes 16-31 -> (M = r+8, N = l).
#pragma unroll
    for (int i = 0; i < 4; ++i) {
        const int rowBase = m0 + i * 16 + hi * 8;
#pragma unroll
        for (int j = 0; j < 4; ++j) {
            const int col = n0 + j * 16 + l;
            float* cp = C + (size_t)rowBase * (size_t)N + col;
#pragma unroll
            for (int r = 0; r < 8; ++r)
                cp[(size_t)r * (size_t)N] = acc[i][j][r];
        }
    }
}

// ----------------- gate * depthwise causal conv3 * gate --------------------
// BCx: [Btot*S, 3H] f32 (B | C | x along last dim).
// y[bs,h] = C[bs,h] * ( w0*Bx[bs-2,h] + w1*Bx[bs-1,h] + w2*Bx[bs,h] ), Bx = B*x,
// with the conv window clipped at each batch's start (left pad 2).
__global__ __launch_bounds__(256)
void conv_gate_kernel(const float* __restrict__ BCx,
                      const float* __restrict__ conv_w,   // [H, 3]
                      __bf16* __restrict__ y,             // [Btot*S, H] bf16
                      int S, int Hd, long total)
{
    long idx = (long)blockIdx.x * blockDim.x + threadIdx.x;
    if (idx >= total) return;
    const int  h  = (int)(idx % Hd);
    const long bs = idx / Hd;
    const int  s  = (int)(bs % S);

    const long rowStride = 3L * Hd;
    const float* row = BCx + bs * rowStride;

    const float w0 = conv_w[h * 3 + 0];
    const float w1 = conv_w[h * 3 + 1];
    const float w2 = conv_w[h * 3 + 2];

    float acc = w2 * row[h] * row[2 * Hd + h];
    if (s >= 1) {
        const float* r1 = row - rowStride;
        acc += w1 * r1[h] * r1[2 * Hd + h];
    }
    if (s >= 2) {
        const float* r2 = row - 2 * rowStride;
        acc += w0 * r2[h] * r2[2 * Hd + h];
    }
    y[idx] = (__bf16)(row[Hd + h] * acc);
}

// ------------------------------- launcher ----------------------------------
extern "C" void kernel_launch(void* const* d_in, const int* in_sizes, int n_in,
                              void* d_out, int out_size, void* d_ws, size_t ws_size,
                              hipStream_t stream) {
    (void)in_sizes; (void)n_in; (void)out_size; (void)ws_size;

    constexpr int  H  = 2048;
    constexpr int  S  = 4096;
    constexpr int  Bt = 2;
    constexpr long M  = (long)Bt * S;      // 8192
    constexpr int  N1 = 3 * H;             // 6144
    constexpr int  N2 = H;                 // 2048
    constexpr int  K  = H;                 // 2048

    const float* hidden = (const float*)d_in[0];  // [2,4096,H]
    const float* W_in   = (const float*)d_in[1];  // [3H,H]
    const float* conv_w = (const float*)d_in[2];  // [H,1,3]
    const float* W_out  = (const float*)d_in[3];  // [H,H]
    float*       out    = (float*)d_out;          // [2,4096,H]

    // Workspace layout (all offsets 256B-aligned)
    char* ws = (char*)d_ws;
    __bf16* hiddenBF = (__bf16*)(ws);                          //  33,554,432 B
    __bf16* WinBF    = (__bf16*)(ws + 33554432);               //  25,165,824 B
    __bf16* WoutBF   = (__bf16*)(ws + 58720256);               //   8,388,608 B
    float*  BCx      = (float*) (ws + 67108864);               // 201,326,592 B
    __bf16* yBF      = (__bf16*)(ws + 268435456);              //  33,554,432 B
                                                               // total ~302 MB

    const int T = 256;

    // 1) f32 -> bf16 conversions
    {
        int n4 = (int)(M * K / 4);          // hidden: 16,777,216 elems
        cvt_f32_to_bf16_x4<<<(n4 + T - 1) / T, T, 0, stream>>>(hidden, hiddenBF, n4);
    }
    {
        int n4 = N1 * K / 4;                // W_in: 12,582,912 elems
        cvt_f32_to_bf16_x4<<<(n4 + T - 1) / T, T, 0, stream>>>(W_in, WinBF, n4);
    }
    {
        int n4 = N2 * K / 4;                // W_out: 4,194,304 elems
        cvt_f32_to_bf16_x4<<<(n4 + T - 1) / T, T, 0, stream>>>(W_out, WoutBF, n4);
    }

    // 2) GEMM1: BCx[M, 3H] = hidden[M, K] @ W_in[3H, K]^T
    {
        dim3 grid((unsigned)(M / 128), (unsigned)(N1 / 256));
        gemm_bf16_wmma<<<grid, T, 0, stream>>>(hiddenBF, WinBF, BCx, K, N1);
    }

    // 3) gating + depthwise causal conv(3) + gating -> y (bf16)
    {
        long total = M * (long)H;           // 16,777,216
        long blocks = (total + T - 1) / T;
        conv_gate_kernel<<<(unsigned)blocks, T, 0, stream>>>(BCx, conv_w, yBF, S, H, total);
    }

    // 4) GEMM2: out[M, H] = y[M, K] @ W_out[H, K]^T
    {
        dim3 grid((unsigned)(M / 128), (unsigned)(N2 / 256));
        gemm_bf16_wmma<<<grid, T, 0, stream>>>(yBF, WoutBF, out, K, N2);
    }
}
